// MihGNNEmbeddingTest1_56633438765543
// MI455X (gfx1250) — matrix operand order repaired and checked by reading, not compiled, not run on hardware
//
#include <hip/hip_runtime.h>
#include <math.h>

// ---------------------------------------------------------------------------
// MI455X / gfx1250, wave32. Pipeline:
//   1) pack embedding (fp32 -> bf16 hi/lo RNE, WMMA B-layout)   [trivial]
//   2) pack W          (fp32 -> bf16 hi/lo RNE, WMMA A-layout)  [trivial]
//   3) aggPart[s] = A_s(rows, ksplit s) @ embedding  via v_wmma_f32_16x16x32_bf16
//      - A_s values {0,.25,.5,.75,2} are exact in bf16 -> truncate via v_perm_b32
//      - split-K x4: 2048 waves so the load pipe can cover the 23.3 TB/s floor
//      - unconditional deep global_prefetch_b8 (speculative, OOB-safe)
//   4) agg = sum of 4 partials (deterministic)
//   5) per-edge MLP + distance via WMMA, per-wave loss partials
//   6) deterministic loss reduction
// ---------------------------------------------------------------------------

typedef __attribute__((ext_vector_type(16))) __bf16          v16bf;
typedef __attribute__((ext_vector_type(16))) unsigned short  v16us;
typedef __attribute__((ext_vector_type(8)))  unsigned int    v8u;
typedef __attribute__((ext_vector_type(8)))  float           v8f;

#define N_NODES 8192
#define D_DIM   32
#define B_EDGES 32768
#define L_LAYERS 4
#define KBLOCKS (N_NODES / 32)          // 256 k-steps of 32
#define KSPLIT  4                       // split-K factor for the big GEMM
#define EDGE_WAVES (B_EDGES / 16)       // 2048 waves, 16 edges each
#define PF_DIST (32 * 8)                // prefetch 8 k-steps (~1KB/lane) ahead

static __device__ __forceinline__ unsigned short f2bf(float x) {
  unsigned int u = __float_as_uint(x);
  u += 0x7FFFu + ((u >> 16) & 1u);      // round-to-nearest-even to bf16
  return (unsigned short)(u >> 16);
}
static __device__ __forceinline__ float bf2f(unsigned short h) {
  return __uint_as_float(((unsigned int)h) << 16);
}
// Pack two f32 -> two bf16 by TRUNCATION in one v_perm_b32.
// (exact when values have <=7 mantissa bits, true for A_s entries)
static __device__ __forceinline__ unsigned int pack_trunc2(float lo, float hi) {
  return __builtin_amdgcn_perm(__float_as_uint(hi), __float_as_uint(lo),
                               0x07060302u);
}
static __device__ __forceinline__ v16bf load_bf16x16(const unsigned short* p) {
  return __builtin_bit_cast(v16bf, *(const v16us*)p);
}
static __device__ __forceinline__ v8f wmma_bf16(v16bf a, v16bf b, v8f c) {
  // D = A(16x32 bf16) * B(32x16 bf16) + C(16x16 f32)
  return __builtin_amdgcn_wmma_f32_16x16x32_bf16(
      /*neg_a=*/false, a, /*neg_b=*/false, b,
      /*c_mod=*/(short)0, c, /*reuse_a=*/false, /*reuse_b=*/false);
}
static __device__ __forceinline__ void split16(const float* v, v16bf& h, v16bf& l) {
  v16us hu, lu;
#pragma unroll
  for (int i = 0; i < 16; ++i) {
    unsigned short hb = f2bf(v[i]);
    hu[i] = hb;
    lu[i] = f2bf(v[i] - bf2f(hb));
  }
  h = __builtin_bit_cast(v16bf, hu);
  l = __builtin_bit_cast(v16bf, lu);
}

// --- pack embedding into WMMA B-matrix layout (32x16 bf16, K = (j<8?j:j+8) + 8*hihalf)
__global__ void pack_e_kernel(const float* __restrict__ emb,
                              unsigned short* __restrict__ ehi,
                              unsigned short* __restrict__ elo) {
  int gid  = blockIdx.x * blockDim.x + threadIdx.x;  // [kb][t][lane][j]
  int j    = gid & 15;
  int lane = (gid >> 4) & 31;
  int t    = (gid >> 9) & 1;
  int kb   = gid >> 10;
  if (kb >= KBLOCKS) return;
  int hb = lane >> 4;
  int n  = (lane & 15) + t * 16;
  int kl = ((j < 8) ? j : (j + 8)) + 8 * hb;
  float v = emb[(size_t)(kb * 32 + kl) * D_DIM + n];
  unsigned short h = f2bf(v);
  ehi[gid] = h;
  elo[gid] = f2bf(v - bf2f(h));
}

// --- pack W into WMMA A-matrix layout (16x32 bf16, K = j + 16*hihalf)
__global__ void pack_w_kernel(const float* __restrict__ W,
                              unsigned short* __restrict__ whi,
                              unsigned short* __restrict__ wlo) {
  int gid = blockIdx.x * blockDim.x + threadIdx.x;   // [l][m][lane][j]
  if (gid >= L_LAYERS * 2 * 32 * 16) return;
  int j    = gid & 15;
  int lane = (gid >> 4) & 31;
  int m    = (gid >> 9) & 1;
  int l    = gid >> 10;
  int hb = lane >> 4;
  int o  = m * 16 + (lane & 15);
  int k  = j + 16 * hb;
  float v = W[((size_t)l * D_DIM + o) * D_DIM + k];
  unsigned short h = f2bf(v);
  whi[gid] = h;
  wlo[gid] = f2bf(v - bf2f(h));
}

// --- split-K GEMM: one wave per (16-row M tile, K split). 2048 waves total.
// A_s streams from HBM exactly once (256 MB); embedding pack stays L2-hot.
__global__ void __launch_bounds__(128)
agg_gemm_kernel(const float* __restrict__ As,
                const unsigned short* __restrict__ ehi,
                const unsigned short* __restrict__ elo,
                float* __restrict__ aggPart) {
  const int lane  = threadIdx.x & 31;
  const int wave  = threadIdx.x >> 5;
  const int wgid  = blockIdx.x * 4 + wave;           // 0..2047
  const int tile  = wgid >> 2;                       // 0..511 (M tile)
  const int split = wgid & 3;                        // 0..3   (K split)
  const int m     = lane & 15;
  const int hb    = lane >> 4;
  const int kb0   = split * (KBLOCKS / KSPLIT);
  const int kb1   = kb0 + (KBLOCKS / KSPLIT);
  const float* arow = As + (size_t)(tile * 16 + m) * N_NODES + 8 * hb;
  v8f acc0 = {};                                     // cols 0..15
  v8f acc1 = {};                                     // cols 16..31
  for (int kb = kb0; kb < kb1; ++kb) {
    const float* ap = arow + kb * 32;
    // per-lane A fragment: two contiguous 32B chunks (K interleave per ISA table)
    float4 a0 = ((const float4*)ap)[0];
    float4 a1 = ((const float4*)ap)[1];
    float4 a2 = ((const float4*)(ap + 16))[0];
    float4 a3 = ((const float4*)(ap + 16))[1];
    // Speculative prefetch: unconditional (OOB-safe, dropped on xlat failure),
    // ~1KB ahead per lane so it leads HBM latency. No EXEC-mask churn.
    __builtin_prefetch(ap + PF_DIST, 0, 1);
    v8u au;                                          // truncating pack: v_perm_b32
    au[0] = pack_trunc2(a0.x, a0.y);
    au[1] = pack_trunc2(a0.z, a0.w);
    au[2] = pack_trunc2(a1.x, a1.y);
    au[3] = pack_trunc2(a1.z, a1.w);
    au[4] = pack_trunc2(a2.x, a2.y);
    au[5] = pack_trunc2(a2.z, a2.w);
    au[6] = pack_trunc2(a3.x, a3.y);
    au[7] = pack_trunc2(a3.z, a3.w);
    v16bf a = __builtin_bit_cast(v16bf, au);         // A_s values exact in bf16
    const unsigned short* eb = ehi + (size_t)kb * 1024 + lane * 16;
    const unsigned short* el = elo + (size_t)kb * 1024 + lane * 16;
    v16bf bh0 = load_bf16x16(eb);
    v16bf bh1 = load_bf16x16(eb + 512);
    v16bf bl0 = load_bf16x16(el);
    v16bf bl1 = load_bf16x16(el + 512);
    acc0 = wmma_bf16(a, bh0, acc0);                  // hi/lo split ~ fp32 accuracy
    acc0 = wmma_bf16(a, bl0, acc0);
    acc1 = wmma_bf16(a, bh1, acc1);
    acc1 = wmma_bf16(a, bl1, acc1);
  }
  const int n = lane & 15;
  float* o0 = aggPart + (size_t)split * N_NODES * D_DIM
            + (size_t)(tile * 16 + 8 * hb) * D_DIM + n;
#pragma unroll
  for (int r = 0; r < 8; ++r) {                      // C/D layout: row = r + 8*hb
    o0[(size_t)r * D_DIM]      = acc0[r];
    o0[(size_t)r * D_DIM + 16] = acc1[r];
  }
}

// --- deterministic combine of the 4 K-split partials
__global__ void combine_kernel(const float* __restrict__ part,
                               float* __restrict__ agg) {
  int i = blockIdx.x * blockDim.x + threadIdx.x;
  if (i < N_NODES * D_DIM) {
    const size_t S = (size_t)N_NODES * D_DIM;
    agg[i] = ((part[i] + part[i + S]) + (part[i + 2 * S] + part[i + 3 * S]));
  }
}

// --- per-edge MLP + distance. One wave handles 16 edges (N dim of WMMA).
__global__ void __launch_bounds__(128)
edge_kernel(const int* __restrict__ edges, const float* __restrict__ labels,
            const float* __restrict__ emb, const float* __restrict__ agg,
            const unsigned short* __restrict__ whi, const unsigned short* __restrict__ wlo,
            const float* __restrict__ bvec, float* __restrict__ out,
            float* __restrict__ lossPartial) {
  const int lane = threadIdx.x & 31;
  const int wave = threadIdx.x >> 5;
  const int wgid = blockIdx.x * 4 + wave;            // 0..2047
  const int n    = lane & 15;
  const int hb   = lane >> 4;
  const int edge = wgid * 16 + n;
  const int src  = edges[edge * 2 + 0];
  const int dst  = edges[edge * 2 + 1];
  const float lab = labels[edge];
  const float* tbl = (lab == 1.0f) ? agg : emb;      // pos edges use aggregated rows

  float sv[16], dv[16];
  {
    const float* p = tbl + (size_t)src * D_DIM + 16 * hb;  // B layout: K = j + 16*hb
    float4 x0 = ((const float4*)p)[0], x1 = ((const float4*)p)[1];
    float4 x2 = ((const float4*)p)[2], x3 = ((const float4*)p)[3];
    sv[0]=x0.x; sv[1]=x0.y; sv[2]=x0.z; sv[3]=x0.w;
    sv[4]=x1.x; sv[5]=x1.y; sv[6]=x1.z; sv[7]=x1.w;
    sv[8]=x2.x; sv[9]=x2.y; sv[10]=x2.z; sv[11]=x2.w;
    sv[12]=x3.x; sv[13]=x3.y; sv[14]=x3.z; sv[15]=x3.w;
  }
  {
    const float* p = tbl + (size_t)dst * D_DIM + 16 * hb;
    float4 x0 = ((const float4*)p)[0], x1 = ((const float4*)p)[1];
    float4 x2 = ((const float4*)p)[2], x3 = ((const float4*)p)[3];
    dv[0]=x0.x; dv[1]=x0.y; dv[2]=x0.z; dv[3]=x0.w;
    dv[4]=x1.x; dv[5]=x1.y; dv[6]=x1.z; dv[7]=x1.w;
    dv[8]=x2.x; dv[9]=x2.y; dv[10]=x2.z; dv[11]=x2.w;
    dv[12]=x3.x; dv[13]=x3.y; dv[14]=x3.z; dv[15]=x3.w;
  }
  v16bf esh, esl, edh, edl;
  split16(sv, esh, esl);
  split16(dv, edh, edl);

  v8f hs[2] = {};
  v8f hd[2] = {};
#pragma unroll
  for (int l = 0; l < L_LAYERS; ++l) {
#pragma unroll
    for (int mm = 0; mm < 2; ++mm) {                 // 2 M-tiles of the 32x32 W_l
      const int widx = ((l * 2 + mm) * 32 + lane) * 16;
      v16bf wh = load_bf16x16(whi + widx);
      v16bf wl = load_bf16x16(wlo + widx);
      v8f t = {};
      t = wmma_bf16(wh, esh, t);                     // 3-term hi/lo product
      t = wmma_bf16(wh, esl, t);
      t = wmma_bf16(wl, esh, t);
      v8f u = {};
      u = wmma_bf16(wh, edh, u);
      u = wmma_bf16(wh, edl, u);
      u = wmma_bf16(wl, edh, u);
      const float* bp = bvec + l * D_DIM + mm * 16 + 8 * hb;
#pragma unroll
      for (int r = 0; r < 8; ++r) {
        float bb = bp[r];
        float ts = fmaxf(t[r] + bb, 0.0f);
        float td = fmaxf(u[r] + bb, 0.0f);
        hs[mm][r] += ts;
        hd[mm][r] += td;
      }
    }
  }
  float p = 0.0f;
#pragma unroll
  for (int mm = 0; mm < 2; ++mm)
#pragma unroll
    for (int r = 0; r < 8; ++r) {
      float df = hs[mm][r] - hd[mm][r];
      p += df * df;
    }
  p += __shfl_xor(p, 16, 32);                        // join the two row-halves
  float dist = p * (1.0f / (float)D_DIM);
  float pred = expf(-dist);
  if (hb == 0) out[1 + edge] = pred;
  float c = (hb == 0) ? 0.5f * (lab - pred) * (lab - pred) : 0.0f;
#pragma unroll
  for (int off = 16; off >= 1; off >>= 1) c += __shfl_down(c, off, 32);
  if (lane == 0) lossPartial[wgid] = c;
}

// --- deterministic loss reduction (no float atomics)
__global__ void loss_reduce_kernel(const float* __restrict__ partial,
                                   float* __restrict__ out) {
  __shared__ float s[256];
  float v = 0.0f;
  for (int i = threadIdx.x; i < EDGE_WAVES; i += 256) v += partial[i];
  s[threadIdx.x] = v;
  __syncthreads();
  for (int off = 128; off >= 1; off >>= 1) {
    if ((int)threadIdx.x < off) s[threadIdx.x] += s[threadIdx.x + off];
    __syncthreads();
  }
  if (threadIdx.x == 0) out[0] = s[0];
}

extern "C" void kernel_launch(void* const* d_in, const int* in_sizes, int n_in,
                              void* d_out, int out_size, void* d_ws, size_t ws_size,
                              hipStream_t stream) {
  (void)in_sizes; (void)n_in; (void)out_size; (void)ws_size;
  const int*   edges  = (const int*)d_in[0];
  const float* labels = (const float*)d_in[1];
  const float* As     = (const float*)d_in[2];
  const float* emb    = (const float*)d_in[3];
  const float* W      = (const float*)d_in[4];
  const float* bvec   = (const float*)d_in[5];
  float* out = (float*)d_out;                        // [0]=loss, [1..B]=predicts

  unsigned char* ws = (unsigned char*)d_ws;
  size_t off = 0;
  float* agg = (float*)(ws + off);     off += (size_t)N_NODES * D_DIM * sizeof(float);
  float* aggPart = (float*)(ws + off); off += (size_t)KSPLIT * N_NODES * D_DIM * sizeof(float);
  unsigned short* ehi = (unsigned short*)(ws + off); off += (size_t)KBLOCKS * 1024 * sizeof(unsigned short);
  unsigned short* elo = (unsigned short*)(ws + off); off += (size_t)KBLOCKS * 1024 * sizeof(unsigned short);
  unsigned short* whi = (unsigned short*)(ws + off); off += (size_t)L_LAYERS * 2 * 32 * 16 * sizeof(unsigned short);
  unsigned short* wlo = (unsigned short*)(ws + off); off += (size_t)L_LAYERS * 2 * 32 * 16 * sizeof(unsigned short);
  float* lossPartial = (float*)(ws + off); off += (size_t)EDGE_WAVES * sizeof(float);

  pack_e_kernel<<<(KBLOCKS * 1024) / 256, 256, 0, stream>>>(emb, ehi, elo);
  pack_w_kernel<<<16, 256, 0, stream>>>(W, whi, wlo);
  // 512 M-tiles x 4 K-splits = 2048 waves = 512 blocks of 4 waves
  agg_gemm_kernel<<<512, 128, 0, stream>>>(As, ehi, elo, aggPart);
  combine_kernel<<<(N_NODES * D_DIM) / 256, 256, 0, stream>>>(aggPart, agg);
  edge_kernel<<<EDGE_WAVES / 4, 128, 0, stream>>>(edges, labels, emb, agg,
                                                  whi, wlo, bvec, out, lossPartial);
  loss_reduce_kernel<<<1, 256, 0, stream>>>(lossPartial, out);
}
